// EvroModel_64235530878979
// MI455X (gfx1250) — matrix operand order repaired and checked by reading, not compiled, not run on hardware
//
#include <hip/hip_runtime.h>
#include <hip/hip_bf16.h>

typedef __attribute__((ext_vector_type(16))) _Float16 v16h;
typedef __attribute__((ext_vector_type(8)))  float    v8f;

#define THREADS 256
#define WAVES 8
#define ROWS_PER_BLOCK 128

// ---- workspace byte offsets ----
#define WS_W1F  0        // 4 ntiles * 8 kblocks * 32 lanes * 16 f16 = 32768 B
#define WS_W2F  32768    // 2 kblocks * 32 * 16 f16 = 2048 B
#define WS_W3F  34816    // 1 frag * 32 * 16 f16 = 1024 B
#define WS_PART 36864    // 1024 float partial sums = 4096 B
#define WS_INV  40960    // 1 float (1/total)

// ---- LDS layout (ushort units) ----
#define XS 264                      // 256 + 8 pad -> bank-conflict-free column gathers
#define XBASE 0
#define XCOUNT (ROWS_PER_BLOCK * XS)        // 33792
#define H1S 72                      // 64 + 8 pad
#define H1BASE XCOUNT
#define H1COUNT (WAVES * 16 * H1S)          // 9216
#define H2S 24                      // 16 + 8 pad
#define H2BASE (H1BASE + H1COUNT)
#define H2COUNT (WAVES * 16 * H2S)          // 3072
#define LDS_TOTAL (H2BASE + H2COUNT)        // 46080 ushorts = 92160 B

union AF { v16h v; unsigned u[8]; };

static __device__ __forceinline__ unsigned short f16bits(float a) {
    _Float16 h = (_Float16)a;
    return __builtin_bit_cast(unsigned short, h);
}
static __device__ __forceinline__ unsigned pack2(float a, float b) {
    return (unsigned)f16bits(a) | ((unsigned)f16bits(b) << 16);
}

// A-matrix (16x32 f16) element mapping, per ISA 7.12.2:
// lane L: m=L&15, h=L>>4 ; vgpr v, half j: k = (v<4?0:16) + h*8 + (v&3)*2 + j
// B-matrix (32x16 f16) element mapping (lanes index N, K in lane-half blocks):
// lane L: n=L&15, h=L>>4 ; vgpr v, half j: k = (v<4?0:16) + h*8 + (v&3)*2 + j (same as A)

// ---------------- weight fragment prep ----------------
__global__ void prep_weights(const float* __restrict__ w1, const float* __restrict__ w2,
                             const float* __restrict__ w3, unsigned short* __restrict__ ws) {
    int idx = blockIdx.x * blockDim.x + threadIdx.x;
    if (idx < 16384) {                       // W1F: [nt=4][kb=8][lane=32][slot=16]
        int frag = idx >> 9, rem = idx & 511;
        int L = rem >> 4, s = rem & 15;
        int v = s >> 1, j = s & 1;
        int nt = frag >> 3, kb = frag & 7;
        int k = kb * 32 + ((v < 4) ? 0 : 16) + (L >> 4) * 8 + (v & 3) * 2 + j;
        int n = nt * 16 + (L & 15);
        ws[WS_W1F / 2 + idx] = f16bits(w1[k * 64 + n]);
    } else if (idx < 16384 + 1024) {         // W2F: [kb=2][lane=32][slot=16]
        int i2 = idx - 16384;
        int frag = i2 >> 9, rem = i2 & 511;
        int L = rem >> 4, s = rem & 15;
        int v = s >> 1, j = s & 1;
        int k = frag * 32 + ((v < 4) ? 0 : 16) + (L >> 4) * 8 + (v & 3) * 2 + j;
        int n = L & 15;
        ws[WS_W2F / 2 + i2] = f16bits(w2[k * 16 + n]);
    } else if (idx < 16384 + 1024 + 512) {   // W3F: padded 32x16, real only k<16, n<4
        int i3 = idx - 16384 - 1024;
        int L = i3 >> 4, s = i3 & 15;
        int v = s >> 1, j = s & 1;
        int k = ((v < 4) ? 0 : 16) + (L >> 4) * 8 + (v & 3) * 2 + j;
        int n = L & 15;
        float val = (k < 16 && n < 4) ? w3[k * 4 + n] : 0.0f;
        ws[WS_W3F / 2 + i3] = f16bits(val);
    }
}

// ---------------- fused MLP + exp + partial sums ----------------
__global__ __launch_bounds__(THREADS) void mlp_kernel(
    const float* __restrict__ x, const float* __restrict__ b1,
    const float* __restrict__ b2, const float* __restrict__ b3,
    const unsigned short* __restrict__ wsu, float* __restrict__ partials,
    float* __restrict__ out) {
    __shared__ unsigned short lds[LDS_TOTAL];
    __shared__ float wsum[WAVES];

    const int tid = threadIdx.x;
    const int lane = tid & 31;
    const int wave = tid >> 5;
    const int row0 = blockIdx.x * ROWS_PER_BLOCK;
    const int m = lane & 15, hl = lane >> 4;

    // ---- stage x tile [128x256] f32 -> f16 into LDS (coalesced b128 loads) ----
    const float4* x4 = (const float4*)x;
    #pragma unroll
    for (int it = 0; it < 32; ++it) {
        int i = tid + THREADS * it;        // 0..8191 float4s of the tile
        int r = i >> 6;                    // 64 float4 per row
        int c4 = i & 63;
        float4 f = x4[(size_t)(row0 + r) * 64 + c4];
        uint2 p;
        p.x = pack2(f.x, f.y);
        p.y = pack2(f.z, f.w);
        *(uint2*)&lds[XBASE + r * XS + c4 * 4] = p;
    }
    __syncthreads();

    const int myrow = wave * 16;

    // ---- build layer-1 A fragments from LDS ----
    AF a[8];
    #pragma unroll
    for (int kb = 0; kb < 8; ++kb) {
        #pragma unroll
        for (int v = 0; v < 8; ++v) {
            int c = kb * 32 + ((v < 4) ? 0 : 16) + hl * 8 + (v & 3) * 2;
            a[kb].u[v] = *(const unsigned*)&lds[XBASE + (myrow + m) * XS + c];
        }
    }

    const v16h* W1F = (const v16h*)(wsu + WS_W1F / 2);
    const v16h* W2F = (const v16h*)(wsu + WS_W2F / 2);
    const v16h* W3F = (const v16h*)(wsu + WS_W3F / 2);

    // ---- layer 1: [16x256]x[256x64], relu, -> LDS h1 (f16) ----
    #pragma unroll
    for (int nt = 0; nt < 4; ++nt) {
        v8f acc = {};
        #pragma unroll
        for (int kb = 0; kb < 8; ++kb) {
            v16h bfr = W1F[(nt * 8 + kb) * 32 + lane];
            acc = __builtin_amdgcn_wmma_f32_16x16x32_f16(
                false, a[kb].v, false, bfr, (short)0, acc, false, false);
        }
        float bias = b1[nt * 16 + m];
        #pragma unroll
        for (int r = 0; r < 8; ++r) {
            float val = acc[r] + bias;
            val = fmaxf(val, 0.0f);
            lds[H1BASE + wave * 16 * H1S + (r + 8 * hl) * H1S + nt * 16 + m] = f16bits(val);
        }
    }
    __syncthreads();

    // ---- layer 2: [16x64]x[64x16], tanh, -> LDS h2 (f16) ----
    AF a2[2];
    #pragma unroll
    for (int kb = 0; kb < 2; ++kb) {
        #pragma unroll
        for (int v = 0; v < 8; ++v) {
            int c = kb * 32 + ((v < 4) ? 0 : 16) + hl * 8 + (v & 3) * 2;
            a2[kb].u[v] = *(const unsigned*)&lds[H1BASE + wave * 16 * H1S + m * H1S + c];
        }
    }
    v8f acc2 = {};
    acc2 = __builtin_amdgcn_wmma_f32_16x16x32_f16(false, a2[0].v, false, W2F[lane],
                                                  (short)0, acc2, false, false);
    acc2 = __builtin_amdgcn_wmma_f32_16x16x32_f16(false, a2[1].v, false, W2F[32 + lane],
                                                  (short)0, acc2, false, false);
    {
        float bias = b2[m];
        #pragma unroll
        for (int r = 0; r < 8; ++r) {
            float t = tanhf(acc2[r] + bias);
            lds[H2BASE + wave * 16 * H2S + (r + 8 * hl) * H2S + m] = f16bits(t);
        }
    }
    __syncthreads();

    // ---- layer 3: [16x16(pad32)]x[16x4(pad 32x16)] ----
    AF a3;
    #pragma unroll
    for (int v = 0; v < 8; ++v) {
        if (v < 4) {
            int c = hl * 8 + (v & 3) * 2;   // k in [0,16)
            a3.u[v] = *(const unsigned*)&lds[H2BASE + wave * 16 * H2S + m * H2S + c];
        } else {
            a3.u[v] = 0u;                   // k >= 16 zero-padded
        }
    }
    v8f acc3 = {};
    acc3 = __builtin_amdgcn_wmma_f32_16x16x32_f16(false, a3.v, false, W3F[lane],
                                                  (short)0, acc3, false, false);

    // ---- epilogue: exp(logit) -> out, deterministic partial sum ----
    float s = 0.0f;
    if (m < 4) {
        float bias = b3[m];
        #pragma unroll
        for (int r = 0; r < 8; ++r) {
            int mm = r + 8 * hl;
            float e = expf(acc3[r] + bias);     // logits bounded by tanh*w3: no max needed
            out[(size_t)(row0 + myrow + mm) * 4 + m] = e;
            s += e;
        }
    }
    #pragma unroll
    for (int off = 16; off > 0; off >>= 1) s += __shfl_xor(s, off, 32);
    if (lane == 0) wsum[wave] = s;
    __syncthreads();
    if (tid == 0) {
        float t = 0.0f;
        #pragma unroll
        for (int w = 0; w < WAVES; ++w) t += wsum[w];
        partials[blockIdx.x] = t;
    }
}

// ---------------- global reduction -> 1/sum ----------------
__global__ void reduce_kernel(const float* __restrict__ partials,
                              float* __restrict__ invOut, int nPart) {
    __shared__ float sm[256];
    float s = 0.0f;
    for (int i = threadIdx.x; i < nPart; i += 256) s += partials[i];
    sm[threadIdx.x] = s;
    __syncthreads();
    for (int off = 128; off > 0; off >>= 1) {
        if ((int)threadIdx.x < off) sm[threadIdx.x] += sm[threadIdx.x + off];
        __syncthreads();
    }
    if (threadIdx.x == 0) *invOut = 1.0f / sm[0];
}

// ---------------- normalize ----------------
__global__ void scale_kernel(float* __restrict__ out, const float* __restrict__ inv, int n4) {
    int i = blockIdx.x * blockDim.x + threadIdx.x;
    if (i < n4) {
        float s = *inv;
        float4* o = (float4*)out;
        float4 v = o[i];
        v.x *= s; v.y *= s; v.z *= s; v.w *= s;
        o[i] = v;
    }
}

extern "C" void kernel_launch(void* const* d_in, const int* in_sizes, int n_in,
                              void* d_out, int out_size, void* d_ws, size_t ws_size,
                              hipStream_t stream) {
    const float* x  = (const float*)d_in[0];
    const float* w1 = (const float*)d_in[1];
    const float* b1 = (const float*)d_in[2];
    const float* w2 = (const float*)d_in[3];
    const float* b2 = (const float*)d_in[4];
    const float* w3 = (const float*)d_in[5];
    const float* b3 = (const float*)d_in[6];
    float* out = (float*)d_out;
    unsigned short* wsu = (unsigned short*)d_ws;
    char* wsb = (char*)d_ws;

    int rows = in_sizes[0] / 256;                 // 131072
    int blocks = rows / ROWS_PER_BLOCK;           // 1024

    prep_weights<<<(16384 + 1024 + 512 + 255) / 256, 256, 0, stream>>>(w1, w2, w3, wsu);
    mlp_kernel<<<blocks, THREADS, 0, stream>>>(x, b1, b2, b3, wsu,
                                               (float*)(wsb + WS_PART), out);
    reduce_kernel<<<1, 256, 0, stream>>>((const float*)(wsb + WS_PART),
                                         (float*)(wsb + WS_INV), blocks);
    scale_kernel<<<(rows + 255) / 256, 256, 0, stream>>>(out, (const float*)(wsb + WS_INV), rows);
}
